// YoloPredict_45311904972954
// MI455X (gfx1250) — compile-verified, matrix operand order checked by reading, not run on hardware
//
#include <hip/hip_runtime.h>
#include <hip/hip_bf16.h>
#include <stdint.h>

typedef __attribute__((ext_vector_type(16))) _Float16 v16h;
typedef __attribute__((ext_vector_type(8)))  float    v8f;

#define NUM_CLASSES 20
#define NUM_ANCHORS 5
#define HH 26
#define WW 26
#define HW 676            // 26*26
#define NBOX 3380         // 5*676
#define BATCH 16
#define CCH 125           // (20+4+1)*5
#define NMS_TH 0.45f
#define PRE_TH 0.005f
#define FEAT_STRIDE 32.0f
#define TILES 43          // ceil(676/16)
#define PROB_ELEMS ((size_t)BATCH * NBOX * (NUM_CLASSES + 1))  // 1,135,680

__constant__ float c_biases[NUM_ANCHORS][2] = {
    {1.08f, 1.19f}, {3.42f, 4.41f}, {6.63f, 11.38f}, {9.42f, 5.11f}, {16.62f, 10.52f}};

__device__ __forceinline__ float sigmoidf_(float v) { return 1.0f / (1.0f + expf(-v)); }

// ---------------------------------------------------------------------------
// Decode kernel: one wave handles 16 spatial positions of one (batch, anchor).
// Softmax denominator computed with v_wmma_f32_16x16x32_f16 (ones x expvals).
// ---------------------------------------------------------------------------
__global__ __launch_bounds__(256)
void yolo_decode(const float* __restrict__ x, const float* __restrict__ im_info,
                 float* __restrict__ prob, float* __restrict__ bbox,
                 float* __restrict__ ws_scores)
{
    int wid  = (int)((blockIdx.x * blockDim.x + threadIdx.x) >> 5);
    int lane = (int)(threadIdx.x & 31);
    if (wid >= BATCH * NUM_ANCHORS * TILES) return;  // wave-uniform

    int b    = wid / (NUM_ANCHORS * TILES);
    int rem  = wid - b * (NUM_ANCHORS * TILES);
    int a    = rem / TILES;
    int tile = rem - a * TILES;

    int col   = lane & 15;    // column n within the 16-wide tile
    int half  = lane >> 4;    // lanes 0-15 supply K=0..15, lanes 16-31 K=16..31
    int hw    = tile * 16 + col;
    bool valid = (hw < HW);
    int hwc   = valid ? hw : 0;

    const float* xb = x + (size_t)b * CCH * HW;

    // ---- class logits for this column; missing slots -> -1e30 (exp -> 0) ----
    float lg[16];
    float mx = -1e30f;
    #pragma unroll
    for (int t = 0; t < 16; ++t) {
        int k = half * 16 + t;
        float v = -1e30f;
        if (k < NUM_CLASSES && valid)
            v = xb[(size_t)(5 * NUM_ANCHORS + k * NUM_ANCHORS + a) * HW + hw];
        lg[t] = v;
        mx = fmaxf(mx, v);
    }
    // combine max between the two half-lanes holding the same column
    mx = fmaxf(mx, __shfl_xor(mx, 16, 32));

    float ev[16];
    v16h bmat;
    #pragma unroll
    for (int t = 0; t < 16; ++t) {
        float e = expf(lg[t] - mx);   // 0 for padded slots
        ev[t] = e;
        bmat[t] = (_Float16)e;
    }
    v16h aones;
    #pragma unroll
    for (int t = 0; t < 16; ++t) aones[t] = (_Float16)1.0f;

    // D[m][n] = sum_k 1 * B[k][n]  ->  every lane gets its column's softmax
    // denominator in c[0] (C/D layout: N = lane%16).
    v8f c = {};
    c = __builtin_amdgcn_wmma_f32_16x16x32_f16(
            /*neg_a=*/false, aones, /*neg_b=*/false, bmat,
            /*c_mod=*/(short)0, c, /*reuse_a=*/false, /*reuse_b=*/false);
    float inv = 1.0f / c[0];

    // ---- objectness ----
    float obj = sigmoidf_(xb[(size_t)(4 * NUM_ANCHORS + a) * HW + hwc]);

    if (valid) {
        size_t n = (size_t)a * HW + hw;
        float* pr = prob + ((size_t)b * NBOX + n) * (NUM_CLASSES + 1);
        #pragma unroll
        for (int t = 0; t < 16; ++t) {
            int k = half * 16 + t;
            if (k < NUM_CLASSES) pr[k] = ev[t] * inv * obj;   // class 20 written by NMS
        }

        if (lane < 16) {
            // ---- box decode (one writer per position) ----
            float sx = sigmoidf_(xb[(size_t)(2 * a + 0) * HW + hw]);
            float sy = sigmoidf_(xb[(size_t)(2 * a + 1) * HW + hw]);
            float w0 = xb[(size_t)(2 * NUM_ANCHORS + 2 * a + 0) * HW + hw];
            float w1 = xb[(size_t)(2 * NUM_ANCHORS + 2 * a + 1) * HW + hw];
            int hrow = hw / WW, wcol = hw - hrow * WW;

            float bx = ((float)wcol + sx) / (float)WW;
            float by = ((float)hrow + sy) / (float)HH;
            float bw = c_biases[a][0] * expf(w0) / (float)WW;
            float bh = c_biases[a][1] * expf(w1) / (float)HH;

            float im_h = im_info[0], im_w = im_info[1];
            float netw = (float)WW * FEAT_STRIDE, neth = (float)HH * FEAT_STRIDE;
            bool cond   = (netw / im_w) < (neth / im_h);
            float new_w = cond ? netw : im_w * neth / im_h;
            float new_h = cond ? im_h * netw / im_w : neth;
            bx = (bx - (netw - new_w) * 0.5f / netw) / (new_w / netw);
            by = (by - (neth - new_h) * 0.5f / neth) / (new_h / neth);
            bw = bw * (netw / new_w);
            bh = bh * (neth / new_h);

            float4 o;
            o.x = bx * im_w; o.y = by * im_h; o.z = bw * im_w; o.w = bh * im_h;
            ((float4*)bbox)[(size_t)b * NBOX + n] = o;
            ws_scores[(size_t)b * NBOX + n] = obj;
        }
    }
}

// ---------------------------------------------------------------------------
// NMS kernel: one 1024-thread workgroup per image; everything lives in LDS.
// Boxes are gathered into LDS in sorted order with gfx1250 async LDS loads.
// ---------------------------------------------------------------------------
#define NSORT 4096
#define NPAD  3392

__global__ __launch_bounds__(1024)
void yolo_nms(const float* __restrict__ bbox, const float* __restrict__ ws_scores,
              float* __restrict__ prob)
{
    __shared__ float  s_key[NSORT];        // 16 KB
    __shared__ int    s_val[NSORT];        // 16 KB
    __shared__ float4 s_box[NPAD];         // 53 KB (x1,y1,x2,y2 after conversion)
    __shared__ unsigned char s_keep[NPAD]; // 3.3 KB
    __shared__ int s_cnt;

    int b   = (int)blockIdx.x;
    int tid = (int)threadIdx.x;

    for (int i = tid; i < NSORT; i += 1024) {
        if (i < NBOX) {
            float s = ws_scores[(size_t)b * NBOX + i];
            s_key[i] = (s < PRE_TH) ? 0.0f : s;
            s_val[i] = i;
        } else { s_key[i] = -1.0f; s_val[i] = i; }   // sentinel sorts last
    }
    if (tid == 0) s_cnt = 0;
    __syncthreads();

    // Bitonic sort: descending by score, ascending index on ties (stable argsort(-s))
    for (int k = 2; k <= NSORT; k <<= 1) {
        for (int j = k >> 1; j > 0; j >>= 1) {
            for (int i = tid; i < NSORT; i += 1024) {
                int p = i ^ j;
                if (p > i) {
                    float ka = s_key[i], kb = s_key[p];
                    int   va = s_val[i], vb = s_val[p];
                    bool before = (ka > kb) || (ka == kb && va < vb);
                    bool desc   = ((i & k) == 0);
                    if (before != desc) {
                        s_key[i] = kb; s_val[i] = vb;
                        s_key[p] = ka; s_val[p] = va;
                    }
                }
            }
            __syncthreads();
        }
    }

    // number of positive (>= PRE_TH) scores = loop bound (sorted descending)
    for (int i = tid; i < NBOX; i += 1024)
        if (s_key[i] > 0.0f) atomicAdd(&s_cnt, 1);

    // gather boxes (cx,cy,w,h) into LDS in sorted order: async global->LDS
    for (int i = tid; i < NBOX; i += 1024) {
        int orig = s_val[i];
        unsigned lds = (unsigned)(uintptr_t)(&s_box[i]);
        unsigned long long ga =
            (unsigned long long)(uintptr_t)(bbox + ((size_t)b * NBOX + orig) * 4);
        asm volatile("global_load_async_to_lds_b128 %0, %1, off"
                     :: "v"(lds), "v"(ga) : "memory");
    }
    asm volatile("s_wait_asynccnt 0x0" ::: "memory");
    __syncthreads();
    int M = s_cnt;

    // convert to corners in place
    for (int i = tid; i < NPAD; i += 1024) {
        if (i < NBOX) {
            float4 v = s_box[i];
            float4 cc;
            cc.x = v.x - v.z * 0.5f; cc.y = v.y - v.w * 0.5f;
            cc.z = v.x + v.z * 0.5f; cc.w = v.y + v.w * 0.5f;
            s_box[i] = cc;
        } else {
            s_box[i] = make_float4(0.f, 0.f, 0.f, 0.f);
        }
        s_keep[i] = 1;
    }
    __syncthreads();

    // greedy suppression (sequential over sorted order, parallel over j)
    for (int i = 0; i < M; ++i) {
        if (s_keep[i]) {                       // uniform branch (LDS broadcast)
            float4 bi = s_box[i];
            float areai = (bi.z - bi.x) * (bi.w - bi.y);
            for (int j = i + 1 + tid; j < NBOX; j += 1024) {
                if (s_keep[j]) {
                    float4 bj = s_box[j];
                    float iw = fminf(bi.z, bj.z) - fmaxf(bi.x, bj.x);
                    float ih = fminf(bi.w, bj.w) - fmaxf(bi.y, bj.y);
                    iw = fmaxf(iw, 0.0f); ih = fmaxf(ih, 0.0f);
                    float inter = iw * ih;
                    float areaj = (bj.z - bj.x) * (bj.w - bj.y);
                    float iou = inter / (areai + areaj - inter + 1e-12f);
                    if (iou > NMS_TH) s_keep[j] = 0;
                }
            }
        }
        __syncthreads();
    }

    // scatter kept scores back to original positions at class index 20
    for (int i = tid; i < NBOX; i += 1024) {
        float v = s_keep[i] ? s_key[i] : 0.0f;
        prob[((size_t)b * NBOX + s_val[i]) * (NUM_CLASSES + 1) + NUM_CLASSES] = v;
    }
}

extern "C" void kernel_launch(void* const* d_in, const int* in_sizes, int n_in,
                              void* d_out, int out_size, void* d_ws, size_t ws_size,
                              hipStream_t stream)
{
    const float* x       = (const float*)d_in[0];
    const float* im_info = (const float*)d_in[1];
    float* prob = (float*)d_out;
    float* bbox = (float*)d_out + PROB_ELEMS;          // bboxs follow prob, flat
    float* ws_scores = (float*)d_ws;                   // 16*3380 floats

    int waves  = BATCH * NUM_ANCHORS * TILES;          // 3440 waves
    int blocks = (waves * 32 + 255) / 256;             // 430 blocks x 8 waves
    yolo_decode<<<blocks, 256, 0, stream>>>(x, im_info, prob, bbox, ws_scores);
    yolo_nms<<<BATCH, 1024, 0, stream>>>(bbox, ws_scores, prob);
}
